// Extractor_27290222199157
// MI455X (gfx1250) — compile-verified, matrix operand order checked by reading, not run on hardware
//
#include <hip/hip_runtime.h>
#include <math.h>

// ---------------------------------------------------------------------------
// Types for CDNA5 WMMA (wave32): v_wmma_f32_16x16x32_f16
// ---------------------------------------------------------------------------
typedef _Float16 half_t;
typedef __attribute__((ext_vector_type(16))) _Float16 v16h;
typedef __attribute__((ext_vector_type(8)))  float    v8f;
typedef __attribute__((ext_vector_type(4)))  unsigned int u32x4;

struct Q2 { u32x4 a, b; };  // 32 bytes == one v16h fragment

// Problem constants
constexpr int  B_    = 16;
constexpr int  S_    = 512;
constexpr int  D_    = 512;
constexpr int  DEPTH = 4;
constexpr int  H_    = 8;
constexpr int  DH_   = 64;
constexpr int  FF_   = 2048;
constexpr int  CBD_  = 64;
constexpr int  CBS_  = 8192;
constexpr long MROWS = (long)B_ * S_;   // 8192

// GEMM tiling: 128x64 tile per 128-thread (4-wave) workgroup, BK=32.
// Each wave owns 32 rows -> 2 A-fragments x 4 B-fragments = 8 WMMA / K-step.
constexpr int BM = 128, BN = 64, BK = 32;
constexpr int KSTRIDE = 40;             // padded LDS row stride (80B, 16B-aligned)

// Epilogue flags
constexpr unsigned EPI_BIAS   = 1u;
constexpr unsigned EPI_POS    = 2u;
constexpr unsigned EPI_RES    = 4u;
constexpr unsigned EPI_GELU   = 8u;
constexpr unsigned EPI_SCALE  = 16u;
constexpr unsigned EPI_OUT32  = 32u;
constexpr unsigned EPI_OUT16  = 64u;
constexpr unsigned EPI_ARGMIN = 128u;
constexpr unsigned EPI_OUT16T = 256u;   // store fp16 transposed: out16[n*ldc + m]

__device__ __forceinline__ unsigned long long shfl_xor_u64(unsigned long long v, int m) {
  unsigned lo = (unsigned)v, hi = (unsigned)(v >> 32);
  lo = __shfl_xor(lo, m, 32);
  hi = __shfl_xor(hi, m, 32);
  return ((unsigned long long)hi << 32) | lo;
}

// CDNA5 async global->LDS copies (ASYNCcnt-tracked). The instruction IOFFSET is
// added to BOTH the LDS destination and the global source (ISA 15.18 / 8.2
// async pseudocode), so consecutive 16B chunks need only immediate offsets.
__device__ __forceinline__ void async4_b128(half_t* lds_dst, const half_t* gsrc) {
  unsigned l = (unsigned)(unsigned long long)lds_dst;
  asm volatile(
      "global_load_async_to_lds_b128 %0, %1, off\n\t"
      "global_load_async_to_lds_b128 %0, %1, off offset:16\n\t"
      "global_load_async_to_lds_b128 %0, %1, off offset:32\n\t"
      "global_load_async_to_lds_b128 %0, %1, off offset:48" ::"v"(l),
      "v"(gsrc)
      : "memory");
}
__device__ __forceinline__ void async2_b128(half_t* lds_dst, const half_t* gsrc) {
  unsigned l = (unsigned)(unsigned long long)lds_dst;
  asm volatile(
      "global_load_async_to_lds_b128 %0, %1, off\n\t"
      "global_load_async_to_lds_b128 %0, %1, off offset:16" ::"v"(l),
      "v"(gsrc)
      : "memory");
}
__device__ __forceinline__ void wait_async0() {
  asm volatile("s_wait_asynccnt 0x0" ::: "memory");
}
// Async loads complete in order: <=6 outstanding means the previous tile's
// 6 copies have landed while the newest 6 stream in behind the WMMAs.
__device__ __forceinline__ void wait_async6() {
  asm volatile("s_wait_asynccnt 0x6" ::: "memory");
}

// ---------------------------------------------------------------------------
// Batched WMMA GEMM: C[M,N] = A[M,K] * B[N,K]^T with fused epilogues.
// B is always given transposed ([N,K] row-major) so staging is contiguous.
// Batch offset per operand: (z/zdiv)*outer + (z%zdiv)*inner.
// Double-buffered async-to-LDS staging pipeline.
// ---------------------------------------------------------------------------
template <unsigned EPI>
__global__ __launch_bounds__(128) void gemm_f16_kernel(
    const half_t* A, int lda, long outerA, long innerA,
    const half_t* Bp, int ldb, long outerB, long innerB,
    float* out32, half_t* out16, int ldc, long outerC, long innerC,
    int K, int zdiv,
    const float* __restrict__ bias,
    const float* __restrict__ pos, int posmod,
    const float* res, int ldres,
    float scale,
    const float* __restrict__ cnorm, unsigned long long* keys) {
  __shared__ __align__(16) half_t As[2 * BM * KSTRIDE];  // 20 KB
  __shared__ __align__(16) half_t Bs[2 * BN * KSTRIDE];  // 10 KB

  const int z = blockIdx.z;
  const long offA = (long)(z / zdiv) * outerA + (long)(z % zdiv) * innerA;
  const long offB = (long)(z / zdiv) * outerB + (long)(z % zdiv) * innerB;
  const long offC = (long)(z / zdiv) * outerC + (long)(z % zdiv) * innerC;
  A  += offA;
  Bp += offB;

  const int m0   = blockIdx.y * BM;
  const int n0   = blockIdx.x * BN;
  const int tid  = threadIdx.x;
  const int wave = tid >> 5;
  const int lane = tid & 31;
  const int r    = lane & 15;
  const int hv   = lane >> 4;

  const v8f vzero = {0.f, 0.f, 0.f, 0.f, 0.f, 0.f, 0.f, 0.f};
  v8f acc[2][4] = {{vzero, vzero, vzero, vzero}, {vzero, vzero, vzero, vzero}};

  // Per-thread staging assignments (fixed across the K loop)
  const half_t* aStage = A + (long)(m0 + tid) * lda;      // one A row / thread
  const int     bn     = tid >> 1;
  const int     bch    = tid & 1;
  const half_t* bStage = Bp + (long)(n0 + bn) * ldb + bch * 16;
  half_t* aDst[2] = {As + tid * KSTRIDE, As + BM * KSTRIDE + tid * KSTRIDE};
  half_t* bDst[2] = {Bs + bn * KSTRIDE + bch * 16,
                     Bs + BN * KSTRIDE + bn * KSTRIDE + bch * 16};

  auto stage = [&](int t) {
    const int sel = t & 1;
    async4_b128(aDst[sel], aStage + t * BK);
    async2_b128(bDst[sel], bStage + t * BK);
  };

  const int nk = K / BK;
  stage(0);  // prologue: tile 0 in flight

  for (int t = 0; t < nk; ++t) {
    if (t + 1 < nk) {
      stage(t + 1);   // issue next tile into the other buffer
      wait_async6();  // tile t has landed; tile t+1 streams behind compute
    } else {
      wait_async0();
    }
    __syncthreads();

    const int sel = t & 1;
    // ---- fragments per the CDNA5 16-bit 16x32 layout:
    // lane (r,hv): elems 0..7 -> K = 8*hv.. ; elems 8..15 -> K = 16+8*hv..
    const half_t* arow0 = As + sel * (BM * KSTRIDE) + (32 * wave + r) * KSTRIDE;
    const half_t* arow1 = arow0 + 16 * KSTRIDE;
    Q2 aq0, aq1;
    aq0.a = *(const u32x4*)(arow0 + 8 * hv);
    aq0.b = *(const u32x4*)(arow0 + 16 + 8 * hv);
    aq1.a = *(const u32x4*)(arow1 + 8 * hv);
    aq1.b = *(const u32x4*)(arow1 + 16 + 8 * hv);
    v16h av0 = __builtin_bit_cast(v16h, aq0);
    v16h av1 = __builtin_bit_cast(v16h, aq1);
#pragma unroll
    for (int nb = 0; nb < 4; ++nb) {
      const half_t* brow = Bs + sel * (BN * KSTRIDE) + (nb * 16 + r) * KSTRIDE;
      Q2 bq;
      bq.a = *(const u32x4*)(brow + 8 * hv);
      bq.b = *(const u32x4*)(brow + 16 + 8 * hv);
      v16h bv = __builtin_bit_cast(v16h, bq);
      acc[0][nb] = __builtin_amdgcn_wmma_f32_16x16x32_f16(
          false, av0, false, bv, (short)0, acc[0][nb], false, false);
      acc[1][nb] = __builtin_amdgcn_wmma_f32_16x16x32_f16(
          false, av1, false, bv, (short)0, acc[1][nb], false, false);
    }
    __syncthreads();  // all reads of buf[sel] done before it is re-staged
  }

  // ---- Epilogue. C/D layout: lane L -> col n = L&15, VGPR i -> row i + 8*(L>>4).
  if (EPI & EPI_ARGMIN) {
#pragma unroll
    for (int f = 0; f < 2; ++f) {
#pragma unroll
      for (int i = 0; i < 8; ++i) {
        const int m = m0 + 32 * wave + 16 * f + 8 * hv + i;
        unsigned long long best = ~0ULL;
#pragma unroll
        for (int nb = 0; nb < 4; ++nb) {
          const int n = n0 + nb * 16 + r;
          float d2 = cnorm[n] - 2.0f * acc[f][nb][i];
          unsigned ub = __float_as_uint(d2);
          ub = (ub & 0x80000000u) ? ~ub : (ub | 0x80000000u);  // monotone map
          unsigned long long key = ((unsigned long long)ub << 32) | (unsigned)n;
          best = key < best ? key : best;
        }
#pragma unroll
        for (int msk = 8; msk >= 1; msk >>= 1) {
          unsigned long long o = shfl_xor_u64(best, msk);
          best = o < best ? o : best;
        }
        if (r == 0) atomicMin(&keys[m], best);
      }
    }
  } else {
#pragma unroll
    for (int f = 0; f < 2; ++f) {
#pragma unroll
      for (int nb = 0; nb < 4; ++nb) {
#pragma unroll
        for (int i = 0; i < 8; ++i) {
          const int m = m0 + 32 * wave + 16 * f + 8 * hv + i;
          const int n = n0 + nb * 16 + r;
          float v = acc[f][nb][i];
          if (EPI & EPI_SCALE) v *= scale;
          if (EPI & EPI_BIAS)  v += bias[n];
          if (EPI & EPI_POS)   v += pos[(long)(m % posmod) * ldc + n];
          if (EPI & EPI_RES)   v += res[(long)m * ldres + n];
          if (EPI & EPI_GELU) {
            float x = v;
            v = 0.5f * x * (1.0f + tanhf(0.7978845608f * (x + 0.044715f * x * x * x)));
          }
          if (EPI & EPI_OUT32)  out32[offC + (long)m * ldc + n] = v;
          if (EPI & EPI_OUT16)  out16[offC + (long)m * ldc + n] = (half_t)v;
          if (EPI & EPI_OUT16T) out16[offC + (long)n * ldc + m] = (half_t)v;
        }
      }
    }
  }
}

// ---------------------------------------------------------------------------
// Tiled transpose + convert: dst[n*K+k] = (f16)src[k*N+n], batch via blockIdx.z
// ---------------------------------------------------------------------------
__global__ __launch_bounds__(256) void transpose_f32_to_f16_kernel(
    const float* __restrict__ src, half_t* __restrict__ dst, int K, int N) {
  __shared__ float tile[32][33];
  const long base = (long)blockIdx.z * K * N;
  const int k0 = blockIdx.y * 32, n0 = blockIdx.x * 32;
  const int tx = threadIdx.x & 31, ty = threadIdx.x >> 5;  // 8 rows / pass
#pragma unroll
  for (int i = 0; i < 32; i += 8)
    tile[ty + i][tx] = src[base + (long)(k0 + ty + i) * N + n0 + tx];
  __syncthreads();
#pragma unroll
  for (int i = 0; i < 32; i += 8)
    dst[base + (long)(n0 + ty + i) * K + k0 + tx] = (half_t)tile[tx][ty + i];
}

// ---------------------------------------------------------------------------
// LayerNorm: fp32 in -> fp16 out, one wave per row (D=512)
// ---------------------------------------------------------------------------
__global__ __launch_bounds__(256) void ln_kernel(
    const float* __restrict__ in, const float* __restrict__ g,
    const float* __restrict__ b, half_t* __restrict__ out, int D, int rows) {
  const int row  = blockIdx.x * 8 + (threadIdx.x >> 5);
  const int lane = threadIdx.x & 31;
  if (row >= rows) return;
  const float* x = in + (long)row * D;
  float s = 0.f, s2 = 0.f;
  for (int j = lane; j < D; j += 32) {
    float v = x[j];
    s += v;
    s2 += v * v;
  }
#pragma unroll
  for (int m = 16; m >= 1; m >>= 1) {
    s  += __shfl_xor(s, m, 32);
    s2 += __shfl_xor(s2, m, 32);
  }
  const float mu  = s / D;
  const float var = s2 / D - mu * mu;
  const float inv = rsqrtf(var + 1e-5f);
  half_t* o = out + (long)row * D;
  for (int j = lane; j < D; j += 32)
    o[j] = (half_t)((x[j] - mu) * inv * g[j] + b[j]);
}

// ---------------------------------------------------------------------------
// Softmax over rows of 512 fp16 values, in place. One wave per row.
// ---------------------------------------------------------------------------
__global__ __launch_bounds__(256) void softmax512_kernel(half_t* data) {
  const long row = (long)blockIdx.x * 8 + (threadIdx.x >> 5);
  const int lane = threadIdx.x & 31;
  half_t* x = data + row * 512;
  float vals[16];
  float mx = -3.0e38f;
#pragma unroll
  for (int j = 0; j < 16; ++j) {
    float v = (float)x[lane + 32 * j];
    vals[j] = v;
    mx = fmaxf(mx, v);
  }
#pragma unroll
  for (int m = 16; m >= 1; m >>= 1) mx = fmaxf(mx, __shfl_xor(mx, m, 32));
  float s = 0.f;
#pragma unroll
  for (int j = 0; j < 16; ++j) {
    vals[j] = __expf(vals[j] - mx);
    s += vals[j];
  }
#pragma unroll
  for (int m = 16; m >= 1; m >>= 1) s += __shfl_xor(s, m, 32);
  const float inv = 1.0f / s;
#pragma unroll
  for (int j = 0; j < 16; ++j) x[lane + 32 * j] = (half_t)(vals[j] * inv);
}

// ---------------------------------------------------------------------------
// Helpers: fp32->fp16 convert, codebook norms, init, finalize
// ---------------------------------------------------------------------------
__global__ void f32_to_f16_kernel(const float* __restrict__ s, half_t* __restrict__ d, long n) {
  long i = (long)blockIdx.x * blockDim.x + threadIdx.x;
  if (i < n) d[i] = (half_t)s[i];
}

__global__ void cnorm_kernel(const float* __restrict__ cb, float* __restrict__ cn, int n, int d) {
  int i = blockIdx.x * blockDim.x + threadIdx.x;
  if (i < n) {
    float s = 0.f;
    for (int j = 0; j < d; ++j) {
      float v = cb[(long)i * d + j];
      s += v * v;
    }
    cn[i] = s;
  }
}

__global__ void init_kernel(unsigned long long* keys, int nkeys, float* out, int nout) {
  int i = blockIdx.x * blockDim.x + threadIdx.x;
  if (i < nkeys) keys[i] = ~0ULL;
  if (i < nout) out[i] = 0.0f;
}

// out[b*64+c] = sum_s codebook[idx[b,s]][c]; out[lossIdx] += mean((q-z)^2)
__global__ __launch_bounds__(64) void finalize_kernel(
    const unsigned long long* __restrict__ keys, const float* __restrict__ cb,
    const float* __restrict__ zF, float* out, int S, int D, int lossIdx) {
  const int b = blockIdx.x;
  const int c = threadIdx.x;  // 64 threads = 2 waves
  float acc = 0.f, loss = 0.f;
  for (int s = 0; s < S; ++s) {
    const int idx = (int)(keys[(long)b * S + s] & 0xFFFFFFFFull);
    const float q = cb[(long)idx * D + c];
    acc += q;
    const float d = q - zF[((long)b * S + s) * D + c];
    loss += d * d;
  }
  out[b * D + c] = acc;
#pragma unroll
  for (int m = 16; m >= 1; m >>= 1) loss += __shfl_xor(loss, m, 32);
  __shared__ float ls[2];
  if ((threadIdx.x & 31) == 0) ls[threadIdx.x >> 5] = loss;
  __syncthreads();
  if (threadIdx.x == 0)
    atomicAdd(&out[lossIdx], (ls[0] + ls[1]) * (1.0f / ((float)B_ * S_ * CBD_)));
}

// ---------------------------------------------------------------------------
// Host orchestration
// ---------------------------------------------------------------------------
extern "C" void kernel_launch(void* const* d_in, const int* in_sizes, int n_in,
                              void* d_out, int out_size, void* d_ws, size_t ws_size,
                              hipStream_t stream) {
  (void)in_sizes; (void)n_in; (void)out_size; (void)ws_size;
  const float* x        = (const float*)d_in[0];
  const float* w_in     = (const float*)d_in[1];
  const float* b_in     = (const float*)d_in[2];
  const float* pos_emb  = (const float*)d_in[3];
  const float* ln1_g    = (const float*)d_in[4];
  const float* ln1_b    = (const float*)d_in[5];
  const float* wq       = (const float*)d_in[6];
  const float* wk       = (const float*)d_in[7];
  const float* wv       = (const float*)d_in[8];
  const float* wo       = (const float*)d_in[9];
  const float* ln2_g    = (const float*)d_in[10];
  const float* ln2_b    = (const float*)d_in[11];
  const float* ff_w1    = (const float*)d_in[12];
  const float* ff_b1    = (const float*)d_in[13];
  const float* ff_w2    = (const float*)d_in[14];
  const float* ff_b2    = (const float*)d_in[15];
  const float* lnf_g    = (const float*)d_in[16];
  const float* lnf_b    = (const float*)d_in[17];
  const float* w_out    = (const float*)d_in[18];
  const float* b_out    = (const float*)d_in[19];
  const float* codebook = (const float*)d_in[20];
  float* out = (float*)d_out;

  // -------- workspace carve (deterministic)
  char* ws = (char*)d_ws;
  size_t off = 0;
  auto alloc = [&](size_t bytes) {
    size_t o = off;
    off += (bytes + 255) & ~(size_t)255;
    return o;
  };
  float*  hF    = (float*)(ws + alloc(MROWS * D_ * 4));           // fp32 residual
  half_t* xH    = (half_t*)(ws + alloc(MROWS * D_ * 2));
  half_t* lnH   = (half_t*)(ws + alloc(MROWS * D_ * 2));
  half_t* qH    = (half_t*)(ws + alloc(MROWS * D_ * 2));
  half_t* kH    = (half_t*)(ws + alloc(MROWS * D_ * 2));
  half_t* vT    = (half_t*)(ws + alloc((size_t)D_ * MROWS * 2));  // V transposed [dh, token]
  half_t* ctxH  = (half_t*)(ws + alloc(MROWS * D_ * 2));
  half_t* bigH  = (half_t*)(ws + alloc((size_t)B_ * H_ * S_ * S_ * 2));  // scores / FF mid
  float*  zF    = (float*)(ws + alloc(MROWS * CBD_ * 4));
  half_t* zH    = (half_t*)(ws + alloc(MROWS * CBD_ * 2));
  unsigned long long* keysW = (unsigned long long*)(ws + alloc(MROWS * 8));
  float*  cnorm = (float*)(ws + alloc((size_t)CBS_ * 4));
  half_t* winT  = (half_t*)(ws + alloc((size_t)D_ * D_ * 2));     // all weights [N,K] f16
  half_t* wqT   = (half_t*)(ws + alloc((size_t)DEPTH * D_ * D_ * 2));
  half_t* wkT   = (half_t*)(ws + alloc((size_t)DEPTH * D_ * D_ * 2));
  half_t* wvT   = (half_t*)(ws + alloc((size_t)DEPTH * D_ * D_ * 2));
  half_t* woT   = (half_t*)(ws + alloc((size_t)DEPTH * D_ * D_ * 2));
  half_t* fw1T  = (half_t*)(ws + alloc((size_t)DEPTH * D_ * FF_ * 2));  // [FF,D] per layer
  half_t* fw2T  = (half_t*)(ws + alloc((size_t)DEPTH * FF_ * D_ * 2));  // [D,FF] per layer
  half_t* woutT = (half_t*)(ws + alloc((size_t)D_ * CBD_ * 2));   // [CBD,D]
  half_t* cbH   = (half_t*)(ws + alloc((size_t)CBS_ * CBD_ * 2)); // already [N,K]

  auto cvt = [&](const float* s, half_t* d, long n) {
    f32_to_f16_kernel<<<dim3((unsigned)((n + 255) / 256)), dim3(256), 0, stream>>>(s, d, n);
  };
  auto tpose = [&](const float* s, half_t* d, int K, int N, int zc) {
    transpose_f32_to_f16_kernel<<<dim3(N / 32, K / 32, zc), 256, 0, stream>>>(s, d, K, N);
  };
  cvt(x, xH, MROWS * D_);
  cvt(codebook, cbH, (long)CBS_ * CBD_);
  tpose(w_in, winT, D_, D_, 1);
  tpose(wq, wqT, D_, D_, DEPTH);
  tpose(wk, wkT, D_, D_, DEPTH);
  tpose(wv, wvT, D_, D_, DEPTH);
  tpose(wo, woT, D_, D_, DEPTH);
  tpose(ff_w1, fw1T, D_, FF_, DEPTH);
  tpose(ff_w2, fw2T, FF_, D_, DEPTH);
  tpose(w_out, woutT, D_, CBD_, 1);
  cnorm_kernel<<<dim3(CBS_ / 256), dim3(256), 0, stream>>>(codebook, cnorm, CBS_, CBD_);
  init_kernel<<<dim3(32), dim3(256), 0, stream>>>(keysW, (int)MROWS, out, B_ * CBD_ + 1);

  const long SD = (long)S_ * D_;  // 262144
  const long SS = (long)S_ * S_;  // 262144
  const dim3 gFull(D_ / BN, MROWS / BM, 1);  // (8, 64)

  // -------- input projection: h = x*Win + b_in + pos
  gemm_f16_kernel<EPI_BIAS | EPI_POS | EPI_OUT32><<<gFull, 128, 0, stream>>>(
      xH, D_, 0, 0, winT, D_, 0, 0, hF, nullptr, D_, 0, 0, D_, 1,
      b_in, pos_emb, S_, nullptr, 0, 1.0f, nullptr, nullptr);

  for (int i = 0; i < DEPTH; ++i) {
    // LN1
    ln_kernel<<<dim3((unsigned)(MROWS / 8)), 256, 0, stream>>>(
        hF, ln1_g + i * D_, ln1_b + i * D_, lnH, D_, (int)MROWS);
    // Q, K projections (normal store), V projection (transposed store [dh,token])
    gemm_f16_kernel<EPI_OUT16><<<gFull, 128, 0, stream>>>(
        lnH, D_, 0, 0, wqT + (long)i * D_ * D_, D_, 0, 0, nullptr, qH, D_, 0, 0, D_, 1,
        nullptr, nullptr, 1, nullptr, 0, 1.0f, nullptr, nullptr);
    gemm_f16_kernel<EPI_OUT16><<<gFull, 128, 0, stream>>>(
        lnH, D_, 0, 0, wkT + (long)i * D_ * D_, D_, 0, 0, nullptr, kH, D_, 0, 0, D_, 1,
        nullptr, nullptr, 1, nullptr, 0, 1.0f, nullptr, nullptr);
    gemm_f16_kernel<EPI_OUT16T><<<gFull, 128, 0, stream>>>(
        lnH, D_, 0, 0, wvT + (long)i * D_ * D_, D_, 0, 0, nullptr, vT, (int)MROWS, 0, 0, D_, 1,
        nullptr, nullptr, 1, nullptr, 0, 1.0f, nullptr, nullptr);
    // scores[b,h] = Q Kᵀ * 1/sqrt(64)  (batched over B*H)
    gemm_f16_kernel<EPI_SCALE | EPI_OUT16><<<dim3(S_ / BN, S_ / BM, B_ * H_), 128, 0, stream>>>(
        qH, D_, SD, DH_, kH, D_, SD, DH_, nullptr, bigH, S_, (long)H_ * SS, SS, DH_, H_,
        nullptr, nullptr, 1, nullptr, 0, 0.125f, nullptr, nullptr);
    // softmax over keys
    softmax512_kernel<<<dim3((unsigned)((long)B_ * H_ * S_ / 8)), 256, 0, stream>>>(bigH);
    // context[b,h] = P V  (B = vT[dh, token], base = h*64 rows, b*512 cols)
    gemm_f16_kernel<EPI_OUT16><<<dim3(DH_ / BN, S_ / BM, B_ * H_), 128, 0, stream>>>(
        bigH, S_, (long)H_ * SS, SS, vT, (int)MROWS, (long)S_, (long)DH_ * MROWS,
        nullptr, ctxH, D_, SD, DH_, S_, H_,
        nullptr, nullptr, 1, nullptr, 0, 1.0f, nullptr, nullptr);
    // h = h + context * Wo
    gemm_f16_kernel<EPI_RES | EPI_OUT32><<<gFull, 128, 0, stream>>>(
        ctxH, D_, 0, 0, woT + (long)i * D_ * D_, D_, 0, 0, hF, nullptr, D_, 0, 0, D_, 1,
        nullptr, nullptr, 1, hF, D_, 1.0f, nullptr, nullptr);
    // LN2
    ln_kernel<<<dim3((unsigned)(MROWS / 8)), 256, 0, stream>>>(
        hF, ln2_g + i * D_, ln2_b + i * D_, lnH, D_, (int)MROWS);
    // FF1 + GELU (bigH reused as the [8192,2048] intermediate)
    gemm_f16_kernel<EPI_BIAS | EPI_GELU | EPI_OUT16>
        <<<dim3(FF_ / BN, MROWS / BM, 1), 128, 0, stream>>>(
            lnH, D_, 0, 0, fw1T + (long)i * D_ * FF_, D_, 0, 0, nullptr, bigH, FF_, 0, 0,
            D_, 1, ff_b1 + i * FF_, nullptr, 1, nullptr, 0, 1.0f, nullptr, nullptr);
    // h = h + g * W2 + b2
    gemm_f16_kernel<EPI_BIAS | EPI_RES | EPI_OUT32><<<gFull, 128, 0, stream>>>(
        bigH, FF_, 0, 0, fw2T + (long)i * FF_ * D_, FF_, 0, 0, hF, nullptr, D_, 0, 0,
        FF_, 1, ff_b2 + i * D_, nullptr, 1, hF, D_, 1.0f, nullptr, nullptr);
  }

  // final LN
  ln_kernel<<<dim3((unsigned)(MROWS / 8)), 256, 0, stream>>>(hF, lnf_g, lnf_b, lnH, D_, (int)MROWS);
  // z = hn * W_out + b_out  (fp32 for the loss, fp16 for the codebook GEMM)
  gemm_f16_kernel<EPI_BIAS | EPI_OUT32 | EPI_OUT16>
      <<<dim3(CBD_ / BN, MROWS / BM, 1), 128, 0, stream>>>(
          lnH, D_, 0, 0, woutT, D_, 0, 0, zF, zH, CBD_, 0, 0, D_, 1,
          b_out, nullptr, 1, nullptr, 0, 1.0f, nullptr, nullptr);
  // codebook search: argmin_n (||c_n||^2 - 2 z.c_n) via WMMA + atomicMin keys
  gemm_f16_kernel<EPI_ARGMIN><<<dim3(CBS_ / BN, MROWS / BM, 1), 128, 0, stream>>>(
      zH, CBD_, 0, 0, cbH, CBD_, 0, 0, nullptr, nullptr, 1, 0, 0, CBD_, 1,
      nullptr, nullptr, 1, nullptr, 0, 1.0f, cnorm, keysW);
  // gather quantized codes, sum over sequence, commitment loss
  finalize_kernel<<<dim3(B_), dim3(CBD_), 0, stream>>>(keysW, codebook, zF, out, S_, CBD_, B_ * CBD_);
}
